// PreCoDualBlock_37452114821655
// MI455X (gfx1250) — matrix operand order repaired and checked by reading, not compiled, not run on hardware
//
#include <hip/hip_runtime.h>
#include <hip/hip_bf16.h>
#include <math.h>

// Problem constants (match reference)
#define BB   4
#define LL   1024
#define DD   512
#define DINV 3072          // EXPAND*D
#define BL   (BB*LL)       // 4096 rows
#define NHD  8
#define HD   64
#define NG   (BB*NHD)      // 32 head-groups
#define QKVN 1544          // 3*D + NH
#define QKVLD 1552         // padded ld (multiple of 16)

typedef __attribute__((ext_vector_type(16))) _Float16 v16h;
typedef __attribute__((ext_vector_type(8)))  _Float16 h8;
typedef __attribute__((ext_vector_type(8)))  float    v8f;

union V16H { v16h v; _Float16 e[16]; };
union H8   { h8   v; _Float16 e[8];  };
union V8F  { v8f  v; float    e[8];  };

// Optional CDNA5 async global->LDS path (guarded; falls back to plain LDS stage)
#if defined(__gfx1250__) && __has_builtin(__builtin_amdgcn_global_load_async_to_lds_b32) && __has_builtin(__builtin_amdgcn_s_wait_asynccnt)
#define ASYNC_LDS 1
#else
#define ASYNC_LDS 0
#endif

__device__ __forceinline__ float wave_sum(float v) {
#pragma unroll
  for (int off = 16; off > 0; off >>= 1) v += __shfl_xor(v, off, 32);
  return v;
}

// ---------------- f16 WMMA fragment loaders ---------------------------------
// A fragment (16x32 f16, row-major, lda in halves): lane&15 = M row;
// halves 0-15 = K[k0+kh .. +15], kh = 16*(lane>=16)
__device__ __forceinline__ v16h load_a16(const _Float16* __restrict__ A, int lda,
                                         int m0, int k0, int lane) {
  int m  = m0 + (lane & 15);
  int kh = (lane >> 4) << 4;
  const _Float16* p = A + (long)m * lda + k0 + kh;
  H8 lo, hi;
  lo.v = *(const h8*)p;
  hi.v = *(const h8*)(p + 8);
  V16H r;
#pragma unroll
  for (int i = 0; i < 8; ++i) { r.e[i] = lo.e[i]; r.e[i + 8] = hi.e[i]; }
  return r.v;
}

// B fragment for C = A * W^T, W row-major [N,K] f16: 16 contiguous halves along K
__device__ __forceinline__ v16h load_b16(const _Float16* __restrict__ W, int ldw,
                                         int n, int k0, int lane) {
  int kb = (lane >> 4) << 4;
  return *(const v16h*)(W + (long)n * ldw + k0 + kb);
}

// Stride-generic f16 loaders (transpose views / batched ops); strides in halves
__device__ __forceinline__ v16h load_a16_gen(const _Float16* __restrict__ A, long sm,
                                             long sk, int m0, int k0, int lane) {
  int m  = m0 + (lane & 15);
  int kh = (lane >> 4) << 4;
  const _Float16* b0 = A + (long)m * sm + (long)(k0 + kh) * sk;
  const _Float16* b1 = b0 + 8 * sk;
  V16H r;
#pragma unroll
  for (int i = 0; i < 8; ++i) { r.e[i] = b0[(long)i * sk]; r.e[i + 8] = b1[(long)i * sk]; }
  return r.v;
}

__device__ __forceinline__ v16h load_b16_gen(const _Float16* __restrict__ B, long sk,
                                             long sn, int n0, int k0, int lane) {
  int n  = n0 + (lane & 15);
  int kb = (lane >> 4) << 4;
  const _Float16* b0 = B + (long)n * sn + (long)(k0 + kb) * sk;
  V16H r;
#pragma unroll
  for (int i = 0; i < 16; ++i) r.e[i] = b0[(long)i * sk];
  return r.v;
}

// ---------------- f32 -> f16 conversion -------------------------------------
__global__ void k_cvt(const float* __restrict__ s, _Float16* __restrict__ d, long n) {
  long i = (long)blockIdx.x * blockDim.x + threadIdx.x;
  if (i < n) d[i] = (_Float16)s[i];
}

// ---------------- WMMA GEMM: C[M,N] = A[M,K](f16) * W[N,K]^T(f16) (+bias) ---
// One wave owns a 16x64 tile: 1 A-fragment feeds 4 WMMAs per k-step.
__global__ void k_gemm16(const _Float16* __restrict__ A, int lda,
                         const _Float16* __restrict__ W, int ldw,
                         const float* __restrict__ bias,
                         float* __restrict__ C32, _Float16* __restrict__ C16,
                         int ldc, int N, int K, int tm, int tn) {
  int wave = (int)((blockIdx.x * blockDim.x + threadIdx.x) >> 5);
  if (wave >= tm * tn) return;
  int lane = threadIdx.x & 31;
  int m0 = (wave / tn) << 4;
  int nb = (wave % tn) << 6;
  int ncl[4];
#pragma unroll
  for (int j = 0; j < 4; ++j) {
    int n = nb + j * 16 + (lane & 15);
    ncl[j] = (n < N) ? n : (N - 1);      // clamp loads on edge tiles
  }
  v8f zero = {0.f, 0.f, 0.f, 0.f, 0.f, 0.f, 0.f, 0.f};
  v8f acc[4];
#pragma unroll
  for (int j = 0; j < 4; ++j) acc[j] = zero;
  for (int k0 = 0; k0 < K; k0 += 32) {
    if (k0 + 64 <= K)  // speculative prefetch of next A k-tile (global_prefetch)
      __builtin_prefetch(A + (long)(m0 + (lane & 15)) * lda + k0 + 32, 0, 1);
    v16h a = load_a16(A, lda, m0, k0, lane);
#pragma unroll
    for (int j = 0; j < 4; ++j) {
      v16h bf = load_b16(W, ldw, ncl[j], k0, lane);
      acc[j] = __builtin_amdgcn_wmma_f32_16x16x32_f16(false, a, false, bf, (short)0,
                                                      acc[j], false, false);
    }
  }
  int rbase = m0 + ((lane >> 4) << 3);
#pragma unroll
  for (int j = 0; j < 4; ++j) {
    int col = nb + j * 16 + (lane & 15);
    if (col < N) {
      float badd = bias ? bias[col] : 0.0f;
      V8F u; u.v = acc[j];
#pragma unroll
      for (int r = 0; r < 8; ++r) {
        float v = u.e[r] + badd;
        long off = (long)(rbase + r) * ldc + col;
        if (C32) C32[off] = v;
        if (C16) C16[off] = (_Float16)v;
      }
    }
  }
}

// ------------- Batched stride-generic WMMA GEMM (Gram / RHS / Z apply) ------
__global__ void k_gemm_bat16(const _Float16* __restrict__ A, long bsA, long sAm, long sAk,
                             const _Float16* __restrict__ Bm, long bsB, long sBk, long sBn,
                             float* __restrict__ C, long bsC, int ldc,
                             int K, int tm, int tn) {
  int g    = blockIdx.y;
  int wave = (int)((blockIdx.x * blockDim.x + threadIdx.x) >> 5);
  if (wave >= tm * tn) return;
  int lane = threadIdx.x & 31;
  int m0 = (wave / tn) << 4;
  int n0 = (wave % tn) << 4;
  const _Float16* Ag = A  + (long)g * bsA;
  const _Float16* Bg = Bm + (long)g * bsB;
  float*          Cg = C  + (long)g * bsC;
  v8f acc = {0.f, 0.f, 0.f, 0.f, 0.f, 0.f, 0.f, 0.f};
  for (int k0 = 0; k0 < K; k0 += 32) {
    v16h a = load_a16_gen(Ag, sAm, sAk, m0, k0, lane);
    v16h b = load_b16_gen(Bg, sBk, sBn, n0, k0, lane);
    acc = __builtin_amdgcn_wmma_f32_16x16x32_f16(false, a, false, b, (short)0, acc,
                                                 false, false);
  }
  int col   = n0 + (lane & 15);
  int rbase = m0 + ((lane >> 4) << 3);
  V8F u; u.v = acc;
#pragma unroll
  for (int r = 0; r < 8; ++r)
    Cg[(long)(rbase + r) * ldc + col] = u.e[r];
}

// ---------------- residual add + RMSNorm -> res (f32) + xn (f16) ------------
__global__ void k_resnorm(const float* __restrict__ x, const float* __restrict__ rsd,
                          const float* __restrict__ w, float* __restrict__ res,
                          _Float16* __restrict__ xn16) {
  int wid = threadIdx.x >> 5, lane = threadIdx.x & 31;
  int row = blockIdx.x * 8 + wid;
  if (row >= BL) return;
  const float* xr = x   + (long)row * DD;
  const float* rr = rsd + (long)row * DD;
  float*     resr = res + (long)row * DD;
  float vals[16]; float ss = 0.f;
#pragma unroll
  for (int i = 0; i < 16; ++i) {
    int j = lane + i * 32;
    float v = xr[j] + rr[j];
    resr[j] = v; vals[i] = v; ss += v * v;
  }
  ss = wave_sum(ss);
  float n   = sqrtf(ss) * 0.04419417382415922f;   // * 512^-0.5
  float inv = 1.0f / (n + 1e-6f);
  _Float16* xnr = xn16 + (long)row * DD;
#pragma unroll
  for (int i = 0; i < 16; ++i) {
    int j = lane + i * 32;
    xnr[j] = (_Float16)(w[j] * vals[i] * inv);
  }
}

// ---------------- causal depthwise conv + bias + silu -> f32 + f16 ----------
__global__ void k_conv_silu(const float* __restrict__ xz, const float* __restrict__ cw,
                            const float* __restrict__ cb, float* __restrict__ xc,
                            _Float16* __restrict__ xc16) {
  long idx = (long)blockIdx.x * blockDim.x + threadIdx.x;
  if (idx >= (long)BL * DINV) return;
  int d   = (int)(idx % DINV);
  int row = (int)(idx / DINV);
  int l   = row & (LL - 1);
  const float* base = xz + (long)row * (2 * DINV) + d;
  float w0 = cw[d * 3 + 0], w1 = cw[d * 3 + 1], w2 = cw[d * 3 + 2];
  float v = w2 * base[0];
  if (l >= 1) v += w1 * base[-(long)(2 * DINV)];
  if (l >= 2) v += w0 * base[-(long)(4 * DINV)];
  v += cb[d];
  float s = v / (1.0f + expf(-v));
  xc[idx]   = s;
  xc16[idx] = (_Float16)s;
}

__global__ void k_ksum(const float* __restrict__ dbl, float* __restrict__ ksum) {
  int row = blockIdx.x * blockDim.x + threadIdx.x;
  if (row >= BL) return;
  const float* p = dbl + (long)row * 48 + 32;
  float s = 0.f;
#pragma unroll
  for (int n = 0; n < 8; ++n) { float k = p[n]; s += k * k; }
  ksum[row] = s;
}

// softplus(dt + b) -> beta = dt/(1+dt*ksum), in place over x-half of xz
__global__ void k_beta(float* __restrict__ xz, const float* __restrict__ dtb,
                       const float* __restrict__ ksum) {
  long idx = (long)blockIdx.x * blockDim.x + threadIdx.x;
  if (idx >= (long)BL * DINV) return;
  int d   = (int)(idx % DINV);
  int row = (int)(idx / DINV);
  float* p = xz + (long)row * (2 * DINV) + d;
  float v  = *p + dtb[d];
  float dt = (v > 20.f) ? v : log1pf(expf(v));
  *p = dt / (1.0f + dt * ksum[row]);
}

// ---------------- sequential Longhorn scan ----------------------------------
// Double-buffered k/q broadcast staged through LDS via async global->LDS copy
// (CDNA5 GLOBAL_LOAD_ASYNC_TO_LDS + s_wait_asynccnt) when available.
__global__ void k_scan(const float* __restrict__ xz,   // beta cols[0,DIN), z cols[DIN,2DIN)
                       const float* __restrict__ xc,   // conv output (f32)
                       _Float16* __restrict__ y16,     // gated output (f16)
                       const float* __restrict__ dbl,  // k,q at cols 32..47
                       const float* __restrict__ Dp) {
  const int chunks = DINV / 256;                        // 12
  int b = blockIdx.x / chunks;
  int d = (blockIdx.x % chunks) * 256 + threadIdx.x;
  __shared__ float skq[2][16];
  float h[8];
#pragma unroll
  for (int n = 0; n < 8; ++n) h[n] = 0.f;
  float Dpd = Dp[d];

  auto stage = [&](int t, int buf) {
    if (threadIdx.x < 16) {
      const float* src = dbl + ((long)b * LL + t) * 48 + 32 + threadIdx.x;
#if ASYNC_LDS
      __builtin_amdgcn_global_load_async_to_lds_b32(
          (__attribute__((address_space(1))) int*)src,
          (__attribute__((address_space(3))) int*)&skq[buf][threadIdx.x], 0, 0);
#else
      skq[buf][threadIdx.x] = *src;
#endif
    }
  };

  stage(0, 0);
  for (int t = 0; t < LL; ++t) {
#if ASYNC_LDS
    __builtin_amdgcn_s_wait_asynccnt(0);
#endif
    __syncthreads();
    float kq[16];
#pragma unroll
    for (int n = 0; n < 16; ++n) kq[n] = skq[t & 1][n];
    if (t + 1 < LL) stage(t + 1, (t + 1) & 1);   // overlap next fetch with compute

    long row = (long)b * LL + t;
    float bt = xz[row * (2 * DINV) + d];
    float xt = xc[row * DINV + d];
    float zt = xz[row * (2 * DINV) + DINV + d];
    float y = 0.f;
#pragma unroll
    for (int n = 0; n < 8; ++n) {
      float kn = kq[n], qn = kq[8 + n];
      h[n] = h[n] * (1.0f - bt * kn * kn) + bt * xt * kn;
      y += h[n] * qn;
    }
    y += Dpd * xt;
    y *= zt / (1.0f + expf(-zt));
    y16[row * DINV + d] = (_Float16)y;
  }
}

// ---------------- head extraction: XK16, tgt16 = XV - XK --------------------
__global__ void k_extract(const float* __restrict__ proj, _Float16* __restrict__ XK16,
                          _Float16* __restrict__ tgt16) {
  long idx = (long)blockIdx.x * blockDim.x + threadIdx.x;
  if (idx >= (long)NG * LL * HD) return;
  int i = (int)(idx % HD);
  long r = idx / HD;
  int l = (int)(r % LL);
  int g = (int)(r / LL);
  int b = g >> 3, hd = g & 7;
  long pbase = ((long)b * LL + l) * QKVLD + hd * HD + i;
  float xk = proj[pbase];
  XK16[idx]  = (_Float16)xk;
  tgt16[idx] = (_Float16)(proj[pbase + DD] - xk);
}

// ---------------- per-head 64x64 solve (G + lam I) X = RHS, in LDS ----------
__global__ void k_solve64(const float* __restrict__ G, float* __restrict__ R) {
  __shared__ float sg[64][65];
  __shared__ float sr[64][65];
  __shared__ float scol[64];
  int g = blockIdx.x, j = threadIdx.x;
  const float* Gg = G + (long)g * 4096;
  float*       Rg = R + (long)g * 4096;
  for (int i = 0; i < 64; ++i) {
    float v = Gg[i * 64 + j];
    if (i == j) v += 1e-6f;
    sg[i][j] = v;
    sr[i][j] = Rg[i * 64 + j];
  }
  __syncthreads();
  for (int k = 0; k < 64; ++k) {             // forward elimination
    scol[j] = sg[j][k];
    __syncthreads();
    float inv = 1.0f / scol[k];
    for (int i = k + 1; i < 64; ++i) {
      float f = scol[i] * inv;
      sg[i][j] -= f * sg[k][j];
      sr[i][j] -= f * sr[k][j];
    }
    __syncthreads();
  }
  for (int k = 63; k >= 0; --k) {            // back substitution
    scol[j] = sg[j][k];
    __syncthreads();
    float xkj = sr[k][j] / scol[k];
    sr[k][j] = xkj;
    for (int i = 0; i < k; ++i) sr[i][j] -= scol[i] * xkj;
    __syncthreads();
  }
  for (int i = 0; i < 64; ++i) Rg[i * 64 + j] = sr[i][j];
}

// ---------------- per-(g,l) LN over 64, scatter into (b,l,D) layout ---------
__global__ void k_ln_head(const float* __restrict__ Z, float* __restrict__ outtmp) {
  int wid = threadIdx.x >> 5, lane = threadIdx.x & 31;
  int r = blockIdx.x * 8 + wid;
  if (r >= NG * LL) return;
  int l = r & (LL - 1);
  int g = r >> 10;
  int b = g >> 3, hd = g & 7;
  const float* zr = Z + (long)r * HD;
  float v0 = zr[lane], v1 = zr[lane + 32];
  float s  = wave_sum(v0 + v1);
  float sq = wave_sum(v0 * v0 + v1 * v1);
  float mu  = s * (1.0f / 64.0f);
  float var = sq * (1.0f / 64.0f) - mu * mu;
  float rs  = rsqrtf(var + 1e-5f);
  float* o = outtmp + ((long)b * LL + l) * DD + hd * HD;
  o[lane]      = (v0 - mu) * rs;
  o[lane + 32] = (v1 - mu) * rs;
}

// ---------------- row LN(512)*pn+pb, times gelu(gate) -> f16 ----------------
__global__ void k_ln_gate(const float* __restrict__ outtmp, const float* __restrict__ proj,
                          const float* __restrict__ pw, const float* __restrict__ pb,
                          _Float16* __restrict__ gated16) {
  int wid = threadIdx.x >> 5, lane = threadIdx.x & 31;
  int row = blockIdx.x * 8 + wid;
  if (row >= BL) return;
  const float* o = outtmp + (long)row * DD;
  float vals[16]; float s = 0.f, sq = 0.f;
#pragma unroll
  for (int i = 0; i < 16; ++i) {
    int j = lane + i * 32;
    float v = o[j]; vals[i] = v; s += v; sq += v * v;
  }
  s = wave_sum(s); sq = wave_sum(sq);
  float mu  = s * (1.0f / 512.0f);
  float var = sq * (1.0f / 512.0f) - mu * mu;
  float rs  = rsqrtf(var + 1e-5f);
  const float* gp = proj + (long)row * QKVLD + 2 * DD;
  _Float16* gr = gated16 + (long)row * DD;
#pragma unroll
  for (int i = 0; i < 16; ++i) {
    int j = lane + i * 32;
    float ln = (vals[i] - mu) * rs * pw[j] + pb[j];
    float gt = gp[j];
    float ge = 0.5f * gt * (1.0f + erff(gt * 0.7071067811865476f));
    gr[j] = (_Float16)(ge * ln);
  }
}

// ---------------- gain blend + outputs --------------------------------------
__global__ void k_final(const float* __restrict__ hred, const float* __restrict__ zl,
                        const float* __restrict__ zt, const float* __restrict__ alpha,
                        const float* __restrict__ beta, float* __restrict__ out,
                        float* __restrict__ gainout) {
  int wid = threadIdx.x >> 5, lane = threadIdx.x & 31;
  int row = blockIdx.x * 8 + wid;
  if (row >= BL) return;
  const float* hr = hred + (long)row * DD;
  float s = 0.f, sq = 0.f;
#pragma unroll
  for (int i = 0; i < 16; ++i) {
    int j = lane + i * 32;
    float v = hr[j]; s += v; sq += v * v;
  }
  s = wave_sum(s); sq = wave_sum(sq);
  float var = (sq - s * s * (1.0f / 512.0f)) * (1.0f / 511.0f);  // ddof=1
  float gsig = 1.0f / (1.0f + expf(-(alpha[0] * var + beta[0])));
  gsig = fminf(fmaxf(gsig, 0.0f), 1.0f);
  if (lane == 0) gainout[row] = gsig;
  const float* zlr = zl + (long)row * DD;
  const float* ztr = zt + (long)row * DD;
  float* orow = out + (long)row * DD;
#pragma unroll
  for (int i = 0; i < 16; ++i) {
    int j = lane + i * 32;
    orow[j] = zlr[j] + gsig * (ztr[j] - zlr[j]);
  }
}

// ============================================================================
extern "C" void kernel_launch(void* const* d_in, const int* in_sizes, int n_in,
                              void* d_out, int out_size, void* d_ws, size_t ws_size,
                              hipStream_t stream) {
  (void)in_sizes; (void)n_in; (void)out_size;
  const float* x         = (const float*)d_in[0];
  const float* residual  = (const float*)d_in[1];
  const float* norm_w    = (const float*)d_in[2];
  const float* in_proj_w = (const float*)d_in[3];
  const float* conv_w    = (const float*)d_in[4];
  const float* conv_b    = (const float*)d_in[5];
  const float* x_proj_w  = (const float*)d_in[6];
  const float* dt_head_w = (const float*)d_in[7];
  const float* dt_head_b = (const float*)d_in[8];
  const float* D_param   = (const float*)d_in[9];
  const float* reduce_w  = (const float*)d_in[10];
  const float* qkv_w     = (const float*)d_in[11];
  const float* qkv_b     = (const float*)d_in[12];
  const float* pn_w      = (const float*)d_in[13];
  const float* pn_b      = (const float*)d_in[14];
  const float* wo_w      = (const float*)d_in[15];
  const float* wo_b      = (const float*)d_in[16];
  const float* qnet_w    = (const float*)d_in[17];
  const float* alpha     = (const float*)d_in[18];
  const float* beta_p    = (const float*)d_in[19];

  float* out_p  = (float*)d_out;
  float* res_p  = out_p + (size_t)BL * DD;
  float* gain_p = out_p + (size_t)2 * BL * DD;

  // ---- scratch carve, 256B aligned slabs ----
  size_t off = 0;
  auto carve32 = [&](size_t n) -> float* {
    float* p = (float*)((char*)d_ws + off);
    off += ((n * 4 + 255) / 256) * 256;
    return p;
  };
  auto carve16 = [&](size_t n) -> _Float16* {
    _Float16* p = (_Float16*)((char*)d_ws + off);
    off += ((n * 2 + 255) / 256) * 256;
    return p;
  };
  // f32 activations
  float* xz     = carve32((size_t)BL * 2 * DINV);   // in_proj out; x-half later = beta
  float* xc     = carve32((size_t)BL * DINV);       // conv+silu (scan input)
  float* dbl    = carve32((size_t)BL * 48);
  float* ksum   = carve32((size_t)BL);
  float* hred   = carve32((size_t)BL * DD);
  float* proj   = carve32((size_t)BL * QKVLD);
  float* Gm     = carve32((size_t)NG * HD * HD);
  float* Wsol   = carve32((size_t)NG * HD * HD);
  float* Zbuf   = carve32((size_t)NG * LL * HD);
  float* outtmp = carve32((size_t)BL * DD);
  float* zlong  = carve32((size_t)BL * DD);
  float* zttt   = carve32((size_t)BL * DD);
  // f16 weight shadows (converted once per launch)
  _Float16* w_inproj = carve16((size_t)2 * DINV * DD);
  _Float16* w_xproj  = carve16((size_t)48 * DINV);
  _Float16* w_dthead = carve16((size_t)DINV * 32);
  _Float16* w_reduce = carve16((size_t)DD * DINV);
  _Float16* w_qkv    = carve16((size_t)QKVN * DD);
  _Float16* w_wo     = carve16((size_t)DD * DD);
  _Float16* w_qnet   = carve16((size_t)DD * DD);
  // f16 activation shadows
  _Float16* xn16    = carve16((size_t)BL * DD);
  _Float16* xc16    = carve16((size_t)BL * DINV);   // conv f16; later = y (in place)
  _Float16* dbl16   = carve16((size_t)BL * 48);
  _Float16* hred16  = carve16((size_t)BL * DD);
  _Float16* gated16 = carve16((size_t)BL * DD);
  _Float16* chid16  = carve16((size_t)BL * DD);
  _Float16* XK16    = carve16((size_t)NG * LL * HD);
  _Float16* tgt16   = carve16((size_t)NG * LL * HD);
  _Float16* Wsol16  = carve16((size_t)NG * HD * HD);
  if (off > ws_size) return;  // insufficient scratch

  auto cvt = [&](const float* s, _Float16* d, long n) {
    k_cvt<<<(int)((n + 255) / 256), 256, 0, stream>>>(s, d, n);
  };
  auto gemm = [&](const _Float16* A, int lda, const _Float16* W, int ldw,
                  const float* bias, float* C32, _Float16* C16, int ldc,
                  int N, int K) {
    int tm = BL / 16, tn = (N + 63) / 64;
    long waves = (long)tm * tn;
    int blocks = (int)((waves * 32 + 255) / 256);
    k_gemm16<<<blocks, 256, 0, stream>>>(A, lda, W, ldw, bias, C32, C16, ldc, N, K,
                                         tm, tn);
  };
  auto bgemm = [&](const _Float16* A, long bsA, long sAm, long sAk,
                   const _Float16* Bp, long bsB, long sBk, long sBn,
                   float* C, long bsC, int ldc, int M, int N, int K) {
    int tm = M / 16, tn = N / 16;
    int blocks = (tm * tn * 32 + 255) / 256;
    dim3 g(blocks, NG);
    k_gemm_bat16<<<g, 256, 0, stream>>>(A, bsA, sAm, sAk, Bp, bsB, sBk, sBn,
                                        C, bsC, ldc, K, tm, tn);
  };

  const long NE = (long)BL * DINV;

  // 0) one-time (per launch) weight conversions f32 -> f16
  cvt(in_proj_w, w_inproj, (long)2 * DINV * DD);
  cvt(x_proj_w,  w_xproj,  (long)48 * DINV);
  cvt(dt_head_w, w_dthead, (long)DINV * 32);
  cvt(reduce_w,  w_reduce, (long)DD * DINV);
  cvt(qkv_w,     w_qkv,    (long)QKVN * DD);
  cvt(wo_w,      w_wo,     (long)DD * DD);
  cvt(qnet_w,    w_qnet,   (long)DD * DD);

  // 1) res = x + residual (to d_out), xn16 = rmsnorm(res)*w
  k_resnorm<<<BL / 8, 256, 0, stream>>>(x, residual, norm_w, res_p, xn16);
  // 2) xz = xn @ in_proj_w^T          (4096 x 6144, K=512)
  gemm(xn16, DD, w_inproj, DD, nullptr, xz, nullptr, 2 * DINV, 2 * DINV, DD);
  // 3) xc = silu(causal_conv(x) + b)  (f32 + f16)
  k_conv_silu<<<(int)((NE + 255) / 256), 256, 0, stream>>>(xz, conv_w, conv_b, xc, xc16);
  // 4) dbl = xc @ x_proj_w^T          (4096 x 48, K=3072)
  gemm(xc16, DINV, w_xproj, DINV, nullptr, dbl, dbl16, 48, 48, DINV);
  // 5) ksum = sum_n k^2
  k_ksum<<<BL / 256, 256, 0, stream>>>(dbl, ksum);
  // 6) dt_raw = dt_low @ dt_head_w^T  (4096 x 3072, K=32) -> x-half of xz
  gemm(dbl16, 48, w_dthead, 32, nullptr, xz, nullptr, 2 * DINV, DINV, 32);
  // 7) beta = softplus(dt+b)/(1+dt*ksum), in place
  k_beta<<<(int)((NE + 255) / 256), 256, 0, stream>>>(xz, dt_head_b, ksum);
  // 8) sequential scan; y (gated, f16) written over xc16
  k_scan<<<BB * (DINV / 256), 256, 0, stream>>>(xz, xc, xc16, dbl, D_param);
  // 9) hred = y @ reduce_w^T          (4096 x 512, K=3072)
  gemm(xc16, DINV, w_reduce, DINV, nullptr, hred, hred16, DD, DD, DINV);
  // 10) proj = hred @ qkv_w^T + b     (4096 x 1544, K=512)
  gemm(hred16, DD, w_qkv, DD, qkv_b, proj, nullptr, QKVLD, QKVN, DD);
  // 11) XK / tgt head extraction (f16)
  k_extract<<<(int)(((long)NG * LL * HD + 255) / 256), 256, 0, stream>>>(proj, XK16, tgt16);
  // 12) G = XK^T XK  (batched 64x64, K=1024; A is transpose view)
  bgemm(XK16, (long)LL * HD, 1, HD, XK16, (long)LL * HD, HD, 1,
        Gm, (long)HD * HD, HD, HD, HD, LL);
  // 13) rhs = XK^T tgt -> Wsol
  bgemm(XK16, (long)LL * HD, 1, HD, tgt16, (long)LL * HD, HD, 1,
        Wsol, (long)HD * HD, HD, HD, HD, LL);
  // 14) solve (G + lam I) W = rhs, per head; then f16 shadow
  k_solve64<<<NG, 64, 0, stream>>>(Gm, Wsol);
  cvt(Wsol, Wsol16, (long)NG * HD * HD);
  // 15) Z = XK @ W (batched 1024x64, K=64)
  bgemm(XK16, (long)LL * HD, HD, 1, Wsol16, (long)HD * HD, HD, 1,
        Zbuf, (long)LL * HD, HD, LL, HD, HD);
  // 16) per-head LN -> (b,l,D) layout
  k_ln_head<<<NG * LL / 8, 256, 0, stream>>>(Zbuf, outtmp);
  // 17) LN(512)*pn + pb, times gelu(gate) -> f16
  k_ln_gate<<<BL / 8, 256, 0, stream>>>(outtmp, proj, pn_w, pn_b, gated16);
  // 18) chid16 = gated @ wo_w^T + b   (f16 only; f32 not needed downstream)
  gemm(gated16, DD, w_wo, DD, wo_b, nullptr, chid16, DD, DD, DD);
  // 19) z_long = hred @ qnet_w^T ; 20) z_ttt = chid @ qnet_w^T
  gemm(hred16, DD, w_qnet, DD, nullptr, zlong, nullptr, DD, DD, DD);
  gemm(chid16, DD, w_qnet, DD, nullptr, zttt, nullptr, DD, DD, DD);
  // 21) gain blend, write out + gain
  k_final<<<BL / 8, 256, 0, stream>>>(hred, zlong, zttt, alpha, beta_p, out_p, gain_p);
}
